// CoAttention_5463198400967
// MI455X (gfx1250) — matrix operand order repaired and checked by reading, not compiled, not run on hardware
//
#include <hip/hip_runtime.h>
#include <hip/hip_bf16.h>

typedef __attribute__((ext_vector_type(16))) __bf16 v16bf;
typedef __attribute__((ext_vector_type(8)))  float  v8f;

union Frag {
  v16bf v;
  unsigned int d[8];
  unsigned short u[16];
};

static constexpr int    B_  = 16;
static constexpr int    L_  = 1024;
static constexpr int    D_  = 256;
static constexpr int    H_  = 256;
static constexpr int    G_  = 1024;   // 4H
static constexpr size_t SOUT = 16ull * 1024 * 1280;   // one sequence's output rows

__device__ __forceinline__ unsigned short f2bf(float f) {
  unsigned int u = __float_as_uint(f);
  u += 0x7FFFu + ((u >> 16) & 1u);
  return (unsigned short)(u >> 16);
}

__device__ __forceinline__ float sigm(float x) { return 1.0f / (1.0f + __expf(-x)); }

// A-fragment: 16x32 bf16 tile, row-major with leading dim ldk.
__device__ __forceinline__ Frag load_a(const unsigned short* __restrict__ base, int ldk, int lane) {
  Frag a;
  int m = lane & 15, kh = lane >> 4;
  const unsigned short* row = base + m * ldk + kh * 8;
  *(uint4*)&a.d[0] = *(const uint4*)(row);
  *(uint4*)&a.d[4] = *(const uint4*)(row + 16);
  return a;
}

// B-fragment "weight style": row n of `base` is output column n, contiguous along K.
__device__ __forceinline__ Frag load_b(const unsigned short* __restrict__ base, int ldk, int lane) {
  Frag b;
  int n = lane & 15, kh = lane >> 4;
  const unsigned short* row = base + n * ldk + kh * 8;
  *(uint4*)&b.d[0] = *(const uint4*)(row);
  *(uint4*)&b.d[4] = *(const uint4*)(row + 16);
  return b;
}

__device__ __forceinline__ v8f wmma_bf16(const Frag& a, const Frag& b, v8f c) {
  return __builtin_amdgcn_wmma_f32_16x16x32_bf16(false, a.v, false, b.v, (short)0, c, false, false);
}

// ---------------------------------------------------------------- casts
__global__ __launch_bounds__(256) void cast_w_kernel(
    const float* __restrict__ wih_f, const float* __restrict__ whh_f,
    const float* __restrict__ bih_f, const float* __restrict__ bhh_f,
    const float* __restrict__ wih_b, const float* __restrict__ whh_b,
    const float* __restrict__ bih_b, const float* __restrict__ bhh_b,
    unsigned short* __restrict__ wih, unsigned short* __restrict__ whh,
    float* __restrict__ bias) {
  int idx = blockIdx.x * 256 + threadIdx.x;
  const int WN = G_ * D_;   // 262144
  if (idx < 2 * WN) {
    int dir = idx / WN, r = idx % WN;
    wih[idx] = f2bf((dir ? wih_b : wih_f)[r]);
  } else if (idx < 4 * WN) {
    int j = idx - 2 * WN;
    int dir = j / WN, r = j % WN;
    whh[j] = f2bf((dir ? whh_b : whh_f)[r]);
  } else if (idx < 4 * WN + 2 * G_) {
    int j = idx - 4 * WN;
    int dir = j / G_, r = j % G_;
    bias[j] = dir ? (bih_b[r] + bhh_b[r]) : (bih_f[r] + bhh_f[r]);
  }
}

__global__ __launch_bounds__(256) void cast_x_kernel(
    const float* __restrict__ x1, const float* __restrict__ x2,
    unsigned short* __restrict__ xbf, float* __restrict__ dout) {
  int idx = blockIdx.x * 256 + threadIdx.x;       // 2*16*1024*256 total
  int s  = idx >> 22;
  int r  = idx & 4194303;
  int d  = r & 255;
  int bl = r >> 8;                                // b*1024 + l
  float v = (s ? x2 : x1)[r];
  xbf[idx] = f2bf(v);
  dout[(size_t)s * SOUT + (size_t)bl * 1280 + d] = v;
}

// ---------------------------------------------------------------- xg = x @ Wih^T + (bih+bhh)
// 2x2 register-blocked: each wave computes a 32x32 patch (4 independent accumulators).
__global__ __launch_bounds__(256) void gemm_xg_kernel(
    const unsigned short* __restrict__ xbf, const unsigned short* __restrict__ wih,
    const float* __restrict__ bias, float* __restrict__ xg) {
  int lane = threadIdx.x & 31, wave = threadIdx.x >> 5;
  int tile = blockIdx.x * 8 + wave;               // 32768 units: 2 dir * 512 mu * 32 nu
  int dir = tile >> 14;
  int rem = tile & 16383;
  int mu = rem >> 5, nu = rem & 31;
  int n = lane & 15, kh = lane >> 4;
  float bv0 = bias[dir * G_ + nu * 32 + n];
  float bv1 = bias[dir * G_ + nu * 32 + 16 + n];
  v8f acc00, acc01, acc10, acc11;
#pragma unroll
  for (int r = 0; r < 8; ++r) { acc00[r] = bv0; acc01[r] = bv1; acc10[r] = bv0; acc11[r] = bv1; }
  const unsigned short* abase = xbf + (size_t)(mu * 32) * D_;
  const unsigned short* bbase = wih + (size_t)dir * G_ * D_ + (size_t)(nu * 32) * D_;
#pragma unroll
  for (int ks = 0; ks < 8; ++ks) {
    Frag a0 = load_a(abase + ks * 32, D_, lane);
    Frag a1 = load_a(abase + (size_t)16 * D_ + ks * 32, D_, lane);
    Frag b0 = load_b(bbase + ks * 32, D_, lane);
    Frag b1 = load_b(bbase + (size_t)16 * D_ + ks * 32, D_, lane);
    acc00 = wmma_bf16(a0, b0, acc00);
    acc01 = wmma_bf16(a0, b1, acc01);
    acc10 = wmma_bf16(a1, b0, acc10);
    acc11 = wmma_bf16(a1, b1, acc11);
  }
  float* out = xg + (size_t)dir * 16384 * G_ + (size_t)(mu * 32) * G_ + nu * 32 + n;
#pragma unroll
  for (int r = 0; r < 8; ++r) {
    size_t ro = (size_t)(r + 8 * kh) * G_;
    out[ro]            = acc00[r];
    out[ro + 16]       = acc01[r];
    out[ro + 16 * G_]  = acc10[r];
    out[ro + 16 * G_ + 16] = acc11[r];
  }
}

// ---------------------------------------------------------------- sequential LSTM recurrence
// One block per direction. 16 waves; wave w owns gate columns {w,w+16,w+32,w+48} (N-tiles)
// so each h-column's i/f/g/o quadruple stays in-wave. h: double-buffered LDS (1 barrier/step);
// c: registers. Next step's xg slice is prefetched (global_prefetch_b8).
__global__ __launch_bounds__(512) void recur_kernel(
    const float* __restrict__ xg,           // [2][16384][1024]
    const unsigned short* __restrict__ whh, // [2][1024][256]
    const float* __restrict__ h0,           // [2][16][256]
    const float* __restrict__ c0,
    unsigned short* __restrict__ obf,       // [16][1024][512] this seq
    float* __restrict__ sout,               // d_out + s*SOUT (row stride 1280)
    float* __restrict__ hn, float* __restrict__ cn) {
  __shared__ unsigned short h_lds[2][16 * 256];
  int tid = threadIdx.x;
  int lane = tid & 31, wave = tid >> 5;
  int dir = blockIdx.x;
  int n = lane & 15, kh = lane >> 4;

  for (int i = tid; i < 4096; i += 512) h_lds[0][i] = f2bf(h0[dir * 4096 + i]);
  v8f cst, hreg;
#pragma unroll
  for (int r = 0; r < 8; ++r) {
    int m = r + 8 * kh;
    cst[r]  = c0[dir * 4096 + m * 256 + wave * 16 + n];
    hreg[r] = 0.f;
  }
  __syncthreads();

  const size_t xg_dir = (size_t)dir * 16384 * 1024;
  const unsigned short* wbase = whh + (size_t)dir * G_ * D_;

  for (int s = 0; s < L_; ++s) {
    int l = dir ? (L_ - 1 - s) : s;
    int rb = s & 1, wb = rb ^ 1;
    // prefetch next step's gate pre-activations (no counter, purely speculative)
    if (s + 1 < L_) {
      int ln = dir ? (L_ - 2 - s) : (s + 1);
#pragma unroll
      for (int q = 0; q < 4; ++q) {
        int col = (wave + 16 * q) * 16 + n;
        __builtin_prefetch(xg + xg_dir + (size_t)ln * 1024 + col + (size_t)(8 * kh) * 1048576, 0, 1);
      }
    }
    v8f acc[4];
#pragma unroll
    for (int q = 0; q < 4; ++q) {
      int col = (wave + 16 * q) * 16 + n;
      const float* xp = xg + xg_dir + (size_t)l * 1024 + col;
#pragma unroll
      for (int r = 0; r < 8; ++r) acc[q][r] = xp[(size_t)(r + 8 * kh) * 1048576];
    }
#pragma unroll
    for (int ks = 0; ks < 8; ++ks) {
      Frag a = load_a(&h_lds[rb][ks * 32], 256, lane);
#pragma unroll
      for (int q = 0; q < 4; ++q) {
        Frag b = load_b(wbase + (size_t)((wave + 16 * q) * 16) * D_ + ks * 32, D_, lane);
        acc[q] = wmma_bf16(a, b, acc[q]);
      }
    }
#pragma unroll
    for (int r = 0; r < 8; ++r) {
      int m = r + 8 * kh;
      float iv = sigm(acc[0][r]);
      float fv = sigm(acc[1][r]);
      float gv = tanhf(acc[2][r]);
      float ov = sigm(acc[3][r]);
      float c = fv * cst[r] + iv * gv;
      cst[r] = c;
      float h = ov * tanhf(c);
      hreg[r] = h;
      int col = wave * 16 + n;
      h_lds[wb][m * 256 + col] = f2bf(h);
      sout[(size_t)(m * 1024 + l) * 1280 + 256 + dir * 256 + col] = h;
      obf[(size_t)(m * 1024 + l) * 512 + dir * 256 + col] = f2bf(h);
    }
    __syncthreads();   // publishes h_lds[wb]; also fences reads of h_lds[rb]
  }
#pragma unroll
  for (int r = 0; r < 8; ++r) {
    int m = r + 8 * kh;
    int col = wave * 16 + n;
    hn[dir * 4096 + m * 256 + col] = hreg[r];
    cn[dir * 4096 + m * 256 + col] = cst[r];
  }
}

// ---------------------------------------------------------------- bf16 transpose o -> oT
__global__ __launch_bounds__(1024) void transpose_kernel(
    const unsigned short* __restrict__ obf, unsigned short* __restrict__ otbf) {
  __shared__ unsigned short t[32][33];
  int tx = threadIdx.x & 31, ty = threadIdx.x >> 5;
  int blk = blockIdx.x;           // 32 sb * 16 kt * 32 jt
  int sb = blk >> 9;
  int rem = blk & 511;
  int kt = rem >> 5, jt = rem & 31;
  size_t base = (size_t)sb * 1024 * 512;
  t[ty][tx] = obf[base + (size_t)(jt * 32 + ty) * 512 + kt * 32 + tx];
  __syncthreads();
  otbf[base + (size_t)(kt * 32 + ty) * 1024 + jt * 32 + tx] = t[tx][ty];
}

// ---------------------------------------------------------------- row norms of o (512-dim)
__global__ __launch_bounds__(256) void norm_kernel(const float* __restrict__ dout,
                                                   float* __restrict__ nrm) {
  int lane = threadIdx.x & 31, wave = threadIdx.x >> 5;
  int row = blockIdx.x * 8 + wave;     // s*16384 + b*1024 + l
  int s = row >> 14, bl = row & 16383;
  const float* p = dout + (size_t)s * SOUT + (size_t)bl * 1280 + 256;
  float acc = 0.f;
#pragma unroll
  for (int t = 0; t < 16; ++t) { float v = p[lane + 32 * t]; acc += v * v; }
#pragma unroll
  for (int off = 16; off > 0; off >>= 1) acc += __shfl_xor(acc, off, 32);
  if (lane == 0) nrm[row] = sqrtf(acc);
}

// ---------------------------------------------------------------- sim = (o1 o2^T) / max(n1 n2, eps)
// 2x2 register-blocked: 32x32 patch per wave.
__global__ __launch_bounds__(256) void dot_kernel(
    const unsigned short* __restrict__ o1, const unsigned short* __restrict__ o2,
    const float* __restrict__ n1, const float* __restrict__ n2,
    float* __restrict__ sim) {
  int lane = threadIdx.x & 31, wave = threadIdx.x >> 5;
  int tile = blockIdx.x * 8 + wave;     // 16 * 32 * 32 units
  int b = tile >> 10;
  int rem = tile & 1023;
  int iu = rem >> 5, ju = rem & 31;
  int n = lane & 15, kh = lane >> 4;
  v8f acc00, acc01, acc10, acc11;
#pragma unroll
  for (int r = 0; r < 8; ++r) { acc00[r] = 0.f; acc01[r] = 0.f; acc10[r] = 0.f; acc11[r] = 0.f; }
  const unsigned short* abase = o1 + (size_t)b * 1024 * 512 + (size_t)(iu * 32) * 512;
  const unsigned short* bbase = o2 + (size_t)b * 1024 * 512 + (size_t)(ju * 32) * 512;
#pragma unroll
  for (int ks = 0; ks < 16; ++ks) {
    Frag a0 = load_a(abase + ks * 32, 512, lane);
    Frag a1 = load_a(abase + (size_t)16 * 512 + ks * 32, 512, lane);
    Frag b0 = load_b(bbase + ks * 32, 512, lane);
    Frag b1 = load_b(bbase + (size_t)16 * 512 + ks * 32, 512, lane);
    acc00 = wmma_bf16(a0, b0, acc00);
    acc01 = wmma_bf16(a0, b1, acc01);
    acc10 = wmma_bf16(a1, b0, acc10);
    acc11 = wmma_bf16(a1, b1, acc11);
  }
  float n2v0 = n2[b * 1024 + ju * 32 + n];
  float n2v1 = n2[b * 1024 + ju * 32 + 16 + n];
  float* out = sim + (size_t)b * 1048576 + (size_t)(iu * 32) * 1024 + ju * 32 + n;
#pragma unroll
  for (int r = 0; r < 8; ++r) {
    int m = r + 8 * kh;
    float n1v0 = n1[b * 1024 + iu * 32 + m];
    float n1v1 = n1[b * 1024 + iu * 32 + 16 + m];
    size_t ro = (size_t)m * 1024;
    out[ro]              = acc00[r] / fmaxf(n1v0 * n2v0, 1e-8f);
    out[ro + 16]         = acc01[r] / fmaxf(n1v0 * n2v1, 1e-8f);
    out[ro + 16 * 1024]  = acc10[r] / fmaxf(n1v1 * n2v0, 1e-8f);
    out[ro + 16 * 1024 + 16] = acc11[r] / fmaxf(n1v1 * n2v1, 1e-8f);
  }
}

// ---------------------------------------------------------------- softmax over rows (axis 2)
__global__ __launch_bounds__(256) void rowsm_kernel(const float* __restrict__ sim,
                                                    unsigned short* __restrict__ p) {
  __shared__ float red[256];
  int tid = threadIdx.x;
  const float* src = sim + (size_t)blockIdx.x * 1024;
  float v[4]; float mx = -3.4e38f;
#pragma unroll
  for (int t = 0; t < 4; ++t) { v[t] = src[tid + 256 * t]; mx = fmaxf(mx, v[t]); }
  red[tid] = mx; __syncthreads();
  for (int off = 128; off > 0; off >>= 1) { if (tid < off) red[tid] = fmaxf(red[tid], red[tid + off]); __syncthreads(); }
  mx = red[0]; __syncthreads();
  float sum = 0.f;
#pragma unroll
  for (int t = 0; t < 4; ++t) { v[t] = __expf(v[t] - mx); sum += v[t]; }
  red[tid] = sum; __syncthreads();
  for (int off = 128; off > 0; off >>= 1) { if (tid < off) red[tid] += red[tid + off]; __syncthreads(); }
  float inv = 1.0f / red[0];
  unsigned short* dst = p + (size_t)blockIdx.x * 1024;
#pragma unroll
  for (int t = 0; t < 4; ++t) dst[tid + 256 * t] = f2bf(v[t] * inv);
}

// ---------------------------------------------------------------- softmax over axis 1, written transposed
__global__ __launch_bounds__(256) void colsm_kernel(const float* __restrict__ sim,
                                                    unsigned short* __restrict__ p2t) {
  __shared__ float red[256];
  int tid = threadIdx.x;
  int b = blockIdx.x >> 10, i = blockIdx.x & 1023;
  const float* base = sim + (size_t)b * 1048576 + i;
  float v[4]; float mx = -3.4e38f;
#pragma unroll
  for (int t = 0; t < 4; ++t) { v[t] = base[(size_t)(tid + 256 * t) * 1024]; mx = fmaxf(mx, v[t]); }
  red[tid] = mx; __syncthreads();
  for (int off = 128; off > 0; off >>= 1) { if (tid < off) red[tid] = fmaxf(red[tid], red[tid + off]); __syncthreads(); }
  mx = red[0]; __syncthreads();
  float sum = 0.f;
#pragma unroll
  for (int t = 0; t < 4; ++t) { v[t] = __expf(v[t] - mx); sum += v[t]; }
  red[tid] = sum; __syncthreads();
  for (int off = 128; off > 0; off >>= 1) { if (tid < off) red[tid] += red[tid + off]; __syncthreads(); }
  float inv = 1.0f / red[0];
  unsigned short* dst = p2t + (size_t)blockIdx.x * 1024;
#pragma unroll
  for (int t = 0; t < 4; ++t) dst[tid + 256 * t] = f2bf(v[t] * inv);
}

// ---------------------------------------------------------------- attentive = P @ O  (via oT)
// 2x2 register-blocked: 32x32 patch per wave.
__global__ __launch_bounds__(256) void att_kernel(
    const unsigned short* __restrict__ p,    // [16][1024][1024]
    const unsigned short* __restrict__ ot,   // [16][512][1024]
    float* __restrict__ sout) {              // rows stride 1280, cols 768..1280
  int lane = threadIdx.x & 31, wave = threadIdx.x >> 5;
  int tile = blockIdx.x * 8 + wave;          // 16 * 32 * 16 units
  int b = tile >> 9;
  int rem = tile & 511;
  int iu = rem >> 4, ku = rem & 15;
  int n = lane & 15, kh = lane >> 4;
  v8f acc00, acc01, acc10, acc11;
#pragma unroll
  for (int r = 0; r < 8; ++r) { acc00[r] = 0.f; acc01[r] = 0.f; acc10[r] = 0.f; acc11[r] = 0.f; }
  const unsigned short* abase = p  + (size_t)b * 1048576 + (size_t)(iu * 32) * 1024;
  const unsigned short* bbase = ot + (size_t)b * 524288  + (size_t)(ku * 32) * 1024;
#pragma unroll 2
  for (int ks = 0; ks < 32; ++ks) {
    Frag a0 = load_a(abase + ks * 32, 1024, lane);
    Frag a1 = load_a(abase + (size_t)16 * 1024 + ks * 32, 1024, lane);
    Frag b0 = load_b(bbase + ks * 32, 1024, lane);
    Frag b1 = load_b(bbase + (size_t)16 * 1024 + ks * 32, 1024, lane);
    acc00 = wmma_bf16(a0, b0, acc00);
    acc01 = wmma_bf16(a0, b1, acc01);
    acc10 = wmma_bf16(a1, b0, acc10);
    acc11 = wmma_bf16(a1, b1, acc11);
  }
  float* out = sout + (size_t)(b * 1024 + iu * 32) * 1280 + 768 + ku * 32 + n;
#pragma unroll
  for (int r = 0; r < 8; ++r) {
    size_t ro = (size_t)(r + 8 * kh) * 1280;
    out[ro]               = acc00[r];
    out[ro + 16]          = acc01[r];
    out[ro + 16 * 1280]   = acc10[r];
    out[ro + 16 * 1280 + 16] = acc11[r];
  }
}

// ---------------------------------------------------------------- host
extern "C" void kernel_launch(void* const* d_in, const int* in_sizes, int n_in,
                              void* d_out, int out_size, void* d_ws, size_t ws_size,
                              hipStream_t stream) {
  const float* x1    = (const float*)d_in[0];
  const float* x2    = (const float*)d_in[1];
  const float* s1_h0 = (const float*)d_in[2];
  const float* s1_c0 = (const float*)d_in[3];
  const float* s2_h0 = (const float*)d_in[4];
  const float* s2_c0 = (const float*)d_in[5];
  const float* Wih_f = (const float*)d_in[6];
  const float* Whh_f = (const float*)d_in[7];
  const float* bih_f = (const float*)d_in[8];
  const float* bhh_f = (const float*)d_in[9];
  const float* Wih_b = (const float*)d_in[10];
  const float* Whh_b = (const float*)d_in[11];
  const float* bih_b = (const float*)d_in[12];
  const float* bhh_b = (const float*)d_in[13];
  float* out = (float*)d_out;

  char* ws = (char*)d_ws;
  size_t off = 0;
  unsigned short* xbf  = (unsigned short*)(ws + off); off += 2ull * 16 * 1024 * 256 * 2;  // 16 MiB
  unsigned short* wih  = (unsigned short*)(ws + off); off += 2ull * 1024 * 256 * 2;       // 1 MiB
  unsigned short* whh  = (unsigned short*)(ws + off); off += 2ull * 1024 * 256 * 2;       // 1 MiB
  float*          bias = (float*)(ws + off);          off += 2ull * 1024 * 4;
  unsigned short* obf  = (unsigned short*)(ws + off); off += 2ull * 16 * 1024 * 512 * 2;  // 32 MiB
  unsigned short* otbf = (unsigned short*)(ws + off); off += 2ull * 16 * 512 * 1024 * 2;  // 32 MiB
  float*          nrm  = (float*)(ws + off);          off += 2ull * 16 * 1024 * 4;
  char* big = ws + off;                               // 128 MiB region (xg, later sim/P1/P2T)
  float*          xg  = (float*)big;
  float*          sim = (float*)big;
  unsigned short* p1  = (unsigned short*)(big + 64ull * 1024 * 1024);
  unsigned short* p2t = (unsigned short*)(big + 96ull * 1024 * 1024);

  float* h1n = out + 2 * SOUT;
  float* c1n = h1n + 8192;
  float* h2n = c1n + 8192;
  float* c2n = h2n + 8192;

  cast_w_kernel<<<(4 * 262144 + 2 * 1024 + 255) / 256, 256, 0, stream>>>(
      Wih_f, Whh_f, bih_f, bhh_f, Wih_b, Whh_b, bih_b, bhh_b, wih, whh, bias);
  cast_x_kernel<<<8388608 / 256, 256, 0, stream>>>(x1, x2, xbf, out);

  // sequence 1
  gemm_xg_kernel<<<4096, 256, 0, stream>>>(xbf, wih, bias, xg);
  recur_kernel<<<2, 512, 0, stream>>>(xg, whh, s1_h0, s1_c0, obf, out, h1n, c1n);
  // sequence 2 (xg buffer reused)
  gemm_xg_kernel<<<4096, 256, 0, stream>>>(xbf + 4194304, wih, bias, xg);
  recur_kernel<<<2, 512, 0, stream>>>(xg, whh, s2_h0, s2_c0, obf + 8388608, out + SOUT, h2n, c2n);

  transpose_kernel<<<16384, 1024, 0, stream>>>(obf, otbf);
  norm_kernel<<<4096, 256, 0, stream>>>(out, nrm);
  dot_kernel<<<2048, 256, 0, stream>>>(obf, obf + 8388608, nrm, nrm + 16384, sim);
  rowsm_kernel<<<16384, 256, 0, stream>>>(sim, p1);
  colsm_kernel<<<16384, 256, 0, stream>>>(sim, p2t);
  att_kernel<<<1024, 256, 0, stream>>>(p1, otbf + 8388608, out);          // s1_att = P1 @ o2
  att_kernel<<<1024, 256, 0, stream>>>(p2t, otbf, out + SOUT);            // s2_att = P2^T @ o1
}
